// VS_60748017435317
// MI455X (gfx1250) — matrix operand order repaired and checked by reading, not compiled, not run on hardware
//
#include <hip/hip_runtime.h>
#include <hip/hip_bf16.h>

// ---------------------------------------------------------------------------
// B=8, S=2048, D=256, A_DIM=64.  All matmuls via v_wmma_f32_16x16x32_bf16.
// ---------------------------------------------------------------------------
typedef __attribute__((ext_vector_type(16))) __bf16 v16bf;
typedef __attribute__((ext_vector_type(8)))  float  v8f;

#define WMMA_BF16(A, Bm, C) \
  __builtin_amdgcn_wmma_f32_16x16x32_bf16(false, (A), false, (Bm), (short)0, (C), false, false)

static __device__ inline v16bf load2(const __bf16* p0, const __bf16* p1) {
  union { v16bf v; uint4 u[2]; } f;
  f.u[0] = *(const uint4*)p0;   // elements 0..7   (16 bytes)
  f.u[1] = *(const uint4*)p1;   // elements 8..15  (16 bytes)
  return f.v;
}

static __device__ inline v8f zero8() {
  v8f z = {0.f, 0.f, 0.f, 0.f, 0.f, 0.f, 0.f, 0.f};
  return z;
}

// ---------------------------------------------------------------------------
// Kernel 1: x (f32, [8,2048,256]) -> x_bf row-major + xT_bf ([8,256,2048])
// ---------------------------------------------------------------------------
__global__ void k_convert(const float* __restrict__ x,
                          __bf16* __restrict__ xb, __bf16* __restrict__ xt) {
  __shared__ float tile[32][33];
  const int b  = blockIdx.z;
  const int s0 = blockIdx.x * 32;
  const int d0 = blockIdx.y * 32;
  const int tx = threadIdx.x, ty = threadIdx.y;   // 32 x 8
  const float* xB = x + (size_t)b * 2048 * 256;
#pragma unroll
  for (int r = 0; r < 4; ++r) {
    int s = s0 + ty + r * 8;
    float v = xB[(size_t)s * 256 + d0 + tx];
    tile[ty + r * 8][tx] = v;
    xb[((size_t)b * 2048 + s) * 256 + d0 + tx] = (__bf16)v;
  }
  __syncthreads();
#pragma unroll
  for (int r = 0; r < 4; ++r) {
    int d = d0 + ty + r * 8;
    xt[((size_t)b * 256 + d) * 2048 + s0 + tx] = (__bf16)tile[tx][ty + r * 8];
  }
}

// ---------------------------------------------------------------------------
// Kernel 2: WT[n][k] bf16, n<64 -> W_mean[k][n], else W_logvar[k][n-64]
// ---------------------------------------------------------------------------
__global__ void k_prep_w(const float* __restrict__ Wm, const float* __restrict__ Wl,
                         __bf16* __restrict__ wt) {
  int idx = blockIdx.x * 256 + threadIdx.x;   // 128*256 = 32768
  int n = idx >> 8;          // 0..127
  int k = idx & 255;         // 0..255
  float v = (n < 64) ? Wm[(size_t)k * 64 + n] : Wl[(size_t)k * 64 + (n - 64)];
  wt[idx] = (__bf16)v;
}

// ---------------------------------------------------------------------------
// Kernel 3: flash attention. One wave = 16 query rows x full D=256 output.
// 4 waves / block, grid = B*S/16/4 = 256 blocks.
// ---------------------------------------------------------------------------
__global__ __launch_bounds__(128, 1)
void k_attn(const __bf16* __restrict__ xb, const __bf16* __restrict__ xt,
            __bf16* __restrict__ adet) {
  __shared__ __align__(16) __bf16 pbuf[4][16 * 40];   // P re-layout scratch (stride 40)
  const int lane = threadIdx.x & 31;
  const int w    = threadIdx.x >> 5;
  const int half = lane >> 4;     // 0|1
  const int l15  = lane & 15;
  const int qt = blockIdx.x * 4 + w;      // 0..1023
  const int b  = qt >> 7;                 // 128 q-tiles per batch
  const int q0 = (qt & 127) << 4;
  const __bf16* xbB = xb + (size_t)b * (2048 * 256);
  const __bf16* xtB = xt + (size_t)b * (256 * 2048);

  // Resident Q fragments (A-layout): lane -> row q0+l15
  v16bf qf[8];
  {
    const __bf16* qr = xbB + (size_t)(q0 + l15) * 256 + half * 8;
#pragma unroll
    for (int i = 0; i < 8; ++i)
      qf[i] = load2(qr + i * 32, qr + i * 32 + 16);
  }

  v8f o[16];
#pragma unroll
  for (int t = 0; t < 16; ++t) o[t] = zero8();
  float m[8], l[8];
#pragma unroll
  for (int j = 0; j < 8; ++j) { m[j] = -INFINITY; l[j] = 0.f; }

  const float SCL = 0.090168440055374f;   // (1/16) * log2(e): softmax in base-2

  for (int kb = 0; kb < 2048; kb += 32) {
    // ---- S = Q K^T over two 16-key subtiles -----------------------------
    v8f s0 = zero8(), s1 = zero8();
    const __bf16* kr0 = xbB + (size_t)(kb + l15) * 256 + half * 8;
    const __bf16* kr1 = kr0 + 16 * 256;
#pragma unroll
    for (int i = 0; i < 8; ++i) {
      v16bf bk0 = load2(kr0 + i * 32, kr0 + i * 32 + 16);
      s0 = WMMA_BF16(qf[i], bk0, s0);
      v16bf bk1 = load2(kr1 + i * 32, kr1 + i * 32 + 16);
      s1 = WMMA_BF16(qf[i], bk1, s1);
    }
    if (kb + 32 < 2048) {                       // hint next K chunk into cache
      __builtin_prefetch(kr0 + 32 * 256, 0, 1);
      __builtin_prefetch(kr1 + 32 * 256, 0, 1);
    }

    // ---- online softmax (rows spread across 16-lane halves) -------------
    float al[8];
#pragma unroll
    for (int j = 0; j < 8; ++j) {
      float a = s0[j] * SCL, c = s1[j] * SCL;
      float r = fmaxf(a, c);
      r = fmaxf(r, __shfl_xor(r, 1, 32));
      r = fmaxf(r, __shfl_xor(r, 2, 32));
      r = fmaxf(r, __shfl_xor(r, 4, 32));
      r = fmaxf(r, __shfl_xor(r, 8, 32));
      float mn = fmaxf(m[j], r);
      float alpha = exp2f(m[j] - mn);
      float p0 = exp2f(a - mn);
      float p1 = exp2f(c - mn);
      float rs = p0 + p1;
      rs += __shfl_xor(rs, 1, 32);
      rs += __shfl_xor(rs, 2, 32);
      rs += __shfl_xor(rs, 4, 32);
      rs += __shfl_xor(rs, 8, 32);
      l[j] = l[j] * alpha + rs;
      m[j] = mn;
      al[j] = alpha;
      s0[j] = p0; s1[j] = p1;
    }
#pragma unroll
    for (int t = 0; t < 16; ++t) {
#pragma unroll
      for (int j = 0; j < 8; ++j) o[t][j] *= al[j];
    }

    // ---- P: C-layout -> A-layout via LDS round trip ---------------------
    __bf16* pw = &pbuf[w][0];
#pragma unroll
    for (int j = 0; j < 8; ++j) {
      pw[(j + half * 8) * 40 + l15]      = (__bf16)s0[j];
      pw[(j + half * 8) * 40 + 16 + l15] = (__bf16)s1[j];
    }
    v16bf pf;
    {
      const __bf16* pr = pw + l15 * 40 + half * 8;
      pf = load2(pr, pr + 16);
    }

    // ---- O += P V  (V columns come from xT: contiguous per-lane loads) --
#pragma unroll
    for (int t = 0; t < 16; ++t) {
      const __bf16* vr = xtB + (size_t)(t * 16 + l15) * 2048 + kb + half * 8;
      v16bf vf = load2(vr, vr + 16);
      o[t] = WMMA_BF16(pf, vf, o[t]);
    }
  }

  // ---- finalize: O /= l, store A_det (bf16) -----------------------------
  float inv[8];
#pragma unroll
  for (int j = 0; j < 8; ++j) inv[j] = 1.0f / l[j];
  __bf16* ar = adet + ((size_t)(b * 2048 + q0 + half * 8)) * 256;
#pragma unroll
  for (int t = 0; t < 16; ++t) {
#pragma unroll
    for (int j = 0; j < 8; ++j)
      ar[(size_t)j * 256 + t * 16 + l15] = (__bf16)(o[t][j] * inv[j]);
  }
}

// ---------------------------------------------------------------------------
// Kernel 4: [16384,256] @ [256,128] (mean|logvar) + bias + reparam epilogue.
// One wave = 16 rows; 4 waves/block; grid = 256 blocks.
// ---------------------------------------------------------------------------
__global__ __launch_bounds__(128, 1)
void k_proj(const __bf16* __restrict__ adet, const __bf16* __restrict__ wt,
            const float* __restrict__ bm, const float* __restrict__ bl,
            const float* __restrict__ eps, float* __restrict__ out) {
  const int lane = threadIdx.x & 31;
  const int w    = threadIdx.x >> 5;
  const int half = lane >> 4;
  const int l15  = lane & 15;
  const int r0 = (blockIdx.x * 4 + w) * 16;   // row tile base, 0..16368

  v16bf af[8];
  const __bf16* ar = adet + (size_t)(r0 + l15) * 256 + half * 8;
#pragma unroll
  for (int i = 0; i < 8; ++i)
    af[i] = load2(ar + i * 32, ar + i * 32 + 16);

  v8f acc[8];
#pragma unroll
  for (int t = 0; t < 8; ++t) acc[t] = zero8();

#pragma unroll
  for (int i = 0; i < 8; ++i) {
#pragma unroll
    for (int t = 0; t < 8; ++t) {
      const __bf16* wr = wt + (size_t)(t * 16 + l15) * 256 + i * 32 + half * 8;
      v16bf wf = load2(wr, wr + 16);
      acc[t] = WMMA_BF16(af[i], wf, acc[t]);
    }
  }

  const size_t N1 = (size_t)8 * 2048 * 64;   // 1048576
#pragma unroll
  for (int t = 0; t < 4; ++t) {
    int a = t * 16 + l15;
    float bmv = bm[a];
    float blv = bl[a];
#pragma unroll
    for (int j = 0; j < 8; ++j) {
      size_t row = (size_t)(r0 + half * 8 + j);
      size_t ro  = row * 64 + a;
      float mean = acc[t][j] + bmv;
      float lg   = acc[t + 4][j] + blv;
      out[ro]          = mean;
      out[N1 + ro]     = lg;
      out[2 * N1 + ro] = mean + expf(0.5f * lg) * eps[ro];
    }
  }
}

// ---------------------------------------------------------------------------
extern "C" void kernel_launch(void* const* d_in, const int* in_sizes, int n_in,
                              void* d_out, int out_size, void* d_ws, size_t ws_size,
                              hipStream_t stream) {
  const float* x   = (const float*)d_in[0];
  const float* Wm  = (const float*)d_in[1];
  const float* bm  = (const float*)d_in[2];
  const float* Wl  = (const float*)d_in[3];
  const float* bl  = (const float*)d_in[4];
  const float* eps = (const float*)d_in[5];
  float* out = (float*)d_out;

  char* ws = (char*)d_ws;
  __bf16* xb   = (__bf16*)(ws);                          // 8 MiB  [8,2048,256]
  __bf16* xt   = (__bf16*)(ws + ((size_t)8  << 20));     // 8 MiB  [8,256,2048]
  __bf16* adet = (__bf16*)(ws + ((size_t)16 << 20));     // 8 MiB  [16384,256]
  __bf16* wt   = (__bf16*)(ws + ((size_t)24 << 20));     // 64 KiB [128,256]

  k_convert<<<dim3(64, 8, 8), dim3(32, 8), 0, stream>>>(x, xb, xt);
  k_prep_w<<<dim3(128), dim3(256), 0, stream>>>(Wm, Wl, wt);
  k_attn<<<dim3(256), dim3(128), 0, stream>>>(xb, xt, adet);
  k_proj<<<dim3(256), dim3(128), 0, stream>>>(adet, wt, bm, bl, eps, out);
}
